// Net_70257075028453
// MI455X (gfx1250) — matrix operand order repaired and checked by reading, not compiled, not run on hardware
//
#include <hip/hip_runtime.h>

#define SNN_B 128
#define SNN_F 700
#define SNN_T 500
#define SNN_H 128
#define SNN_O 20
#define SNN_BETA 0.95f

typedef __attribute__((ext_vector_type(16))) __bf16 v16bf;
typedef __attribute__((ext_vector_type(2)))  __bf16 v2bf;
typedef __attribute__((ext_vector_type(8)))  float  v8f;

__device__ __forceinline__ unsigned short f2bf(float f) {
  unsigned int u = __float_as_uint(f);
  u += 0x7FFFu + ((u >> 16) & 1u);   // round-to-nearest-even
  return (unsigned short)(u >> 16);
}

// Pack two f32 -> packed bf16x2 dword. Single v_cvt_pk_bf16_f32 when available.
#if __has_builtin(__builtin_amdgcn_cvt_pk_bf16_f32)
__device__ __forceinline__ unsigned int pack_bf16(float x0, float x1) {
  union { v2bf v; unsigned int u; } c;
  c.v = __builtin_amdgcn_cvt_pk_bf16_f32(x0, x1);
  return c.u;
}
#else
__device__ __forceinline__ unsigned int pack_bf16(float x0, float x1) {
  return (unsigned int)f2bf(x0) | ((unsigned int)f2bf(x1) << 16);
}
#endif

__device__ __forceinline__ v8f wmma_bf16(v16bf a, v16bf b, v8f c) {
  return __builtin_amdgcn_wmma_f32_16x16x32_bf16(false, a, false, b, (short)0, c,
                                                 false, false);
}

// ---------------------------------------------------------------------------
// Swizzle W1 [H,F] (f32) -> bf16 dense-B fragments.
// frag = kt*8 + nt   (kt: 22 f-tiles of 32, nt: 8 h-tiles of 16)
// Within a frag, lane L holds 16 bf16; element i -> K = (i&7) + ((i>>3)<<4) + 8*(L>>4), N = L&15.
// ---------------------------------------------------------------------------
__global__ void swz_w1(const float* __restrict__ W1, unsigned short* __restrict__ w1s) {
  int gid = blockIdx.x * blockDim.x + threadIdx.x;
  if (gid >= 22 * 8 * 512) return;
  int frag = gid >> 9, rem = gid & 511;
  int lane = rem >> 4, i = rem & 15;
  int kt = frag >> 3, nt = frag & 7;
  int k = (i & 7) + ((i >> 3) << 4) + 8 * (lane >> 4);
  int f = kt * 32 + k;
  int h = nt * 16 + (lane & 15);
  float v = (f < SNN_F) ? W1[h * SNN_F + f] : 0.f;
  w1s[gid] = f2bf(v);
}

// W2 [O,H] (f32) -> bf16 dense-B fragments. frag = kt*2 + nt (kt: 4 h-tiles,
// nt: 2 o-tiles), o >= 20 zero-padded.
__global__ void swz_w2(const float* __restrict__ W2, unsigned short* __restrict__ w2s) {
  int gid = blockIdx.x * blockDim.x + threadIdx.x;
  if (gid >= 8 * 512) return;
  int frag = gid >> 9, rem = gid & 511;
  int lane = rem >> 4, i = rem & 15;
  int kt = frag >> 1, nt = frag & 1;
  int k = (i & 7) + ((i >> 3) << 4) + 8 * (lane >> 4);
  int h = kt * 32 + k;
  int o = nt * 16 + (lane & 15);
  float v = (o < SNN_O) ? W2[o * SNN_H + h] : 0.f;
  w2s[gid] = f2bf(v);
}

// ---------------------------------------------------------------------------
// Layer-1 GEMM: cur1[t,b,h] = sum_f X[b,f,t]*W1[h,f] + b1[h]
// Per wave: M = 16 t-rows, N = full H = 128 (8 accumulators), one batch b.
// Block = 8 waves -> 128 t per block. Grid = (B=128, ceil(T/128)=4).
// A-fragment layout (16-bit A 16x32, ISA 7.12.2): lane = row, element i ->
// K = (i&7) + ((i>>3)<<4) + 8*(lane>>4).
// K-loop split: 21 full tiles with no bounds checks + 1 predicated tail (F=700).
// ---------------------------------------------------------------------------
__global__ void __launch_bounds__(256) snn_gemm1(const float* __restrict__ X,
                                                 const unsigned short* __restrict__ w1s,
                                                 const float* __restrict__ b1,
                                                 float* __restrict__ cur1) {
  const int lane = threadIdx.x & 31;
  const int wave = threadIdx.x >> 5;
  const int bidx = blockIdx.x;
  const int t0 = blockIdx.y * 128 + wave * 16;
  const int m = lane & 15;
  const int hi = lane >> 4;

  int trow = t0 + m;
  if (trow > SNN_T - 1) trow = SNN_T - 1;   // clamped rows are never stored
  const float* xb = X + (size_t)bidx * SNN_F * SNN_T + trow;

  v8f acc[8];
#pragma unroll
  for (int nt = 0; nt < 8; nt++) acc[nt] = (v8f){0.f,0.f,0.f,0.f,0.f,0.f,0.f,0.f};

  // -------- 21 full k-tiles, no bounds checks --------
  for (int kt = 0; kt < 21; kt++) {
    const int f0 = kt * 32;
    union { v16bf v; unsigned int u[8]; } a;
#pragma unroll
    for (int p = 0; p < 8; p++) {
      int i0 = 2 * p, i1 = 2 * p + 1;
      int k0 = (i0 & 7) + ((i0 >> 3) << 4) + 8 * hi;
      int k1 = (i1 & 7) + ((i1 >> 3) << 4) + 8 * hi;
      float x0 = xb[(f0 + k0) * SNN_T];
      float x1 = xb[(f0 + k1) * SNN_T];
      a.u[p] = pack_bf16(x0, x1);
    }
    const v16bf* bw = (const v16bf*)(w1s + kt * 8 * 512);
    v16bf bv[8];
#pragma unroll
    for (int nt = 0; nt < 8; nt++) bv[nt] = bw[nt * 32 + lane];
#pragma unroll
    for (int nt = 0; nt < 8; nt++) acc[nt] = wmma_bf16(a.v, bv[nt], acc[nt]);
  }

  // -------- tail k-tile (f0 = 672, 28 valid f) --------
  {
    const int f0 = 21 * 32;
    union { v16bf v; unsigned int u[8]; } a;
#pragma unroll
    for (int p = 0; p < 8; p++) {
      int i0 = 2 * p, i1 = 2 * p + 1;
      int k0 = (i0 & 7) + ((i0 >> 3) << 4) + 8 * hi;
      int k1 = (i1 & 7) + ((i1 >> 3) << 4) + 8 * hi;
      int fa = f0 + k0, fb = f0 + k1;
      float x0 = (fa < SNN_F) ? xb[fa * SNN_T] : 0.f;
      float x1 = (fb < SNN_F) ? xb[fb * SNN_T] : 0.f;
      a.u[p] = pack_bf16(x0, x1);
    }
    const v16bf* bw = (const v16bf*)(w1s + 21 * 8 * 512);
    v16bf bv[8];
#pragma unroll
    for (int nt = 0; nt < 8; nt++) bv[nt] = bw[nt * 32 + lane];
#pragma unroll
    for (int nt = 0; nt < 8; nt++) acc[nt] = wmma_bf16(a.v, bv[nt], acc[nt]);
  }

  // C layout: vgpr r, lane L -> (m = r + 8*(L>>4), n = L&15)
#pragma unroll
  for (int nt = 0; nt < 8; nt++) {
    float bias = b1[nt * 16 + (lane & 15)];
#pragma unroll
    for (int r = 0; r < 8; r++) {
      int t = t0 + r + 8 * hi;
      if (t < SNN_T)
        cur1[((size_t)t * SNN_B + bidx) * SNN_H + nt * 16 + (lane & 15)] =
            acc[nt][r] + bias;
    }
  }
}

// ---------------------------------------------------------------------------
// Temporal scan. 8 blocks x 256 threads; block owns 16 batches.
// mem1 in registers (8 elems/thread), spikes staged in LDS as bf16 A-fragments
// (double-buffered), wave 0 does the 16x128x20 layer-2 GEMM with 8 WMMAs and
// holds mem2/spk2 in accumulator-layout registers across the whole scan.
// ---------------------------------------------------------------------------
__global__ void __launch_bounds__(256) snn_scan(const float* __restrict__ cur1,
                                                const unsigned short* __restrict__ w2s,
                                                const float* __restrict__ b2,
                                                const float* __restrict__ thr1p,
                                                const float* __restrict__ thr2p,
                                                float* __restrict__ out) {
  const int tid = threadIdx.x;
  const int lane = tid & 31;
  const int wave = tid >> 5;
  const int b0 = blockIdx.x * 16;
  const float thr1 = thr1p[0];
  const float thr2 = thr2p[0];

  __shared__ __attribute__((aligned(32))) unsigned short spk[2][2048];

  const int b = tid & 15;
  const int h0 = (tid >> 4) * 8;

  float mem1[8], rst1[8];
#pragma unroll
  for (int j = 0; j < 8; j++) { mem1[j] = 0.f; rst1[j] = 0.f; }

  // LDS offsets matching the 16-bit A-fragment layout: for (row b, K=h):
  // lane = b + 16*((k>>3)&1), element i = (k&7) | (((k>>4)&1)<<3)
  int loff[8];
#pragma unroll
  for (int j = 0; j < 8; j++) {
    int h = h0 + j, kt = h >> 5, k = h & 31;
    int l = b + (((k >> 3) & 1) << 4);
    int i = (k & 7) | (((k >> 4) & 1) << 3);
    loff[j] = kt * 512 + l * 16 + i;
  }

  v16bf w2f[8];
  v8f mem2a = (v8f){0.f,0.f,0.f,0.f,0.f,0.f,0.f,0.f};
  v8f mem2b = mem2a, s2a = mem2a, s2b = mem2a;
  float bia = 0.f, bib = 0.f;
  if (wave == 0) {
#pragma unroll
    for (int j = 0; j < 8; j++) w2f[j] = ((const v16bf*)w2s)[j * 32 + lane];
    int n = lane & 15;
    bia = b2[n];                       // o = n, always < 20
    bib = (n < 4) ? b2[16 + n] : 0.f;  // o = 16+n
  }

  const size_t TBO = (size_t)SNN_T * SNN_B * SNN_O;

  for (int t = 0; t < SNN_T; t++) {
    const float* cp = cur1 + ((size_t)t * SNN_B + (b0 + b)) * SNN_H + h0;
    float4 c0 = ((const float4*)cp)[0];
    float4 c1 = ((const float4*)cp)[1];
    if (t + 1 < SNN_T) __builtin_prefetch(cp + SNN_B * SNN_H, 0, 0);
    float cur[8] = {c0.x, c0.y, c0.z, c0.w, c1.x, c1.y, c1.z, c1.w};

    unsigned short* sb = spk[t & 1];
#pragma unroll
    for (int j = 0; j < 8; j++) {
      float mm = SNN_BETA * mem1[j] + cur[j] - rst1[j] * thr1;
      mem1[j] = mm;
      float s = (mm - thr1 > 0.f) ? 1.f : 0.f;
      rst1[j] = s;
      sb[loff[j]] = (s != 0.f) ? (unsigned short)0x3F80 : (unsigned short)0;
    }
    __syncthreads();

    if (wave == 0) {
      v16bf av0 = ((const v16bf*)(sb +    0))[lane];
      v16bf av1 = ((const v16bf*)(sb +  512))[lane];
      v16bf av2 = ((const v16bf*)(sb + 1024))[lane];
      v16bf av3 = ((const v16bf*)(sb + 1536))[lane];
      v8f ca = (v8f){0.f,0.f,0.f,0.f,0.f,0.f,0.f,0.f};
      v8f cb = ca;
      ca = wmma_bf16(av0, w2f[0], ca);  cb = wmma_bf16(av0, w2f[1], cb);
      ca = wmma_bf16(av1, w2f[2], ca);  cb = wmma_bf16(av1, w2f[3], cb);
      ca = wmma_bf16(av2, w2f[4], ca);  cb = wmma_bf16(av2, w2f[5], cb);
      ca = wmma_bf16(av3, w2f[6], ca);  cb = wmma_bf16(av3, w2f[7], cb);

      int n = lane & 15;
      size_t base = (size_t)t * (SNN_B * SNN_O) +
                    (size_t)(b0 + ((lane >> 4) << 3)) * SNN_O;
#pragma unroll
      for (int r = 0; r < 8; r++) {
        float mm = SNN_BETA * mem2a[r] + (ca[r] + bia) - s2a[r] * thr2;
        mem2a[r] = mm;
        float s = (mm - thr2 > 0.f) ? 1.f : 0.f;
        s2a[r] = s;
        out[base + r * SNN_O + n] = mm;            // mem_rec, o = n (< 16)
        out[TBO + base + r * SNN_O + n] = s;       // spk_rec
      }
      if (n < 4) {
#pragma unroll
        for (int r = 0; r < 8; r++) {
          float mm = SNN_BETA * mem2b[r] + (cb[r] + bib) - s2b[r] * thr2;
          mem2b[r] = mm;
          float s = (mm - thr2 > 0.f) ? 1.f : 0.f;
          s2b[r] = s;
          out[base + r * SNN_O + 16 + n] = mm;     // o = 16..19
          out[TBO + base + r * SNN_O + 16 + n] = s;
        }
      }
    }
  }
}

// ---------------------------------------------------------------------------
extern "C" void kernel_launch(void* const* d_in, const int* in_sizes, int n_in,
                              void* d_out, int out_size, void* d_ws, size_t ws_size,
                              hipStream_t stream) {
  const float* X    = (const float*)d_in[0];
  const float* W1   = (const float*)d_in[1];
  const float* b1   = (const float*)d_in[2];
  const float* thr1 = (const float*)d_in[3];
  const float* W2   = (const float*)d_in[4];
  const float* b2   = (const float*)d_in[5];
  const float* thr2 = (const float*)d_in[6];
  float* out = (float*)d_out;

  char* ws = (char*)d_ws;
  float* cur1          = (float*)ws;                                  // 32,768,000 B
  unsigned short* w1s  = (unsigned short*)(ws + 32768000);            //    180,224 B
  unsigned short* w2s  = (unsigned short*)(ws + 32768000 + 180224);   //      8,192 B

  hipLaunchKernelGGL(swz_w1, dim3((22 * 8 * 512 + 255) / 256), dim3(256), 0, stream,
                     W1, w1s);
  hipLaunchKernelGGL(swz_w2, dim3((8 * 512 + 255) / 256), dim3(256), 0, stream,
                     W2, w2s);
  hipLaunchKernelGGL(snn_gemm1, dim3(SNN_B, 4), dim3(256), 0, stream,
                     X, w1s, b1, cur1);
  hipLaunchKernelGGL(snn_scan, dim3(SNN_B / 16), dim3(256), 0, stream,
                     cur1, w2s, b2, thr1, thr2, out);
}
